// CFGSubASTExpressionCombiner_51007031607312
// MI455X (gfx1250) — matrix operand order repaired and checked by reading, not compiled, not run on hardware
//
#include <hip/hip_runtime.h>
#include <hip/hip_bf16.h>

typedef __attribute__((ext_vector_type(16))) __bf16 v16bf;
typedef __attribute__((ext_vector_type(8)))  float  v8f;

#define DD 256
#define LDS_STRIDE 272   // 256 + 16 pad: keeps 32B alignment, breaks bank periodicity

// ---------------- init workspace ----------------
__global__ void init_ws_kernel(float* __restrict__ acc, float* __restrict__ denom,
                               unsigned* __restrict__ segmax, int* __restrict__ qidx, int P) {
    int g = blockIdx.x * blockDim.x + threadIdx.x;
    long total = (long)P * DD;
    if (g < total) acc[g] = 0.0f;
    if (g < P) {
        denom[g]  = 0.0f;
        segmax[g] = 0xFF800000u;   // -inf bits
        qidx[g]   = -1;
    }
}

__global__ void scatter_qidx_kernel(const int* __restrict__ key, const int* __restrict__ val,
                                    int* __restrict__ qidx, int P) {
    int p = blockIdx.x * blockDim.x + threadIdx.x;
    if (p < P) {
        int k = key[p];
        if (k >= 0 && k < P) qidx[k] = val[p];
    }
}

// Zero qk rows whose query was never set (qidx < 0); keeps WMMA inner loop mask-free.
__global__ void fix_unset_rows_kernel(float* __restrict__ qk, const int* __restrict__ qidx, int P) {
    int g = blockIdx.x * blockDim.x + threadIdx.x;   // one thread per element
    long total = (long)P * DD;
    if (g < total && qidx[g >> 8] < 0) qk[g] = 0.0f;
}

// ---------------- bf16 WMMA GEMM: C[M,256] = gather(A,row_idx) @ (W or W^T) ----------------
// block = 256 threads = 8 waves; each wave owns one 16x16 output tile.
// grid.x covers 128 rows per block, grid.y = 16 column tiles.
__global__ void gemm_bf16_wmma_kernel(const float* __restrict__ A,
                                      const int* __restrict__ row_idx,   // nullptr => identity
                                      const float* __restrict__ W,       // [256,256]
                                      float* __restrict__ C,             // [M,256]
                                      int M, int transposeW) {
    __shared__ __bf16 lw[16 * LDS_STRIDE];   // B slab for this col-tile, [n][k] bf16

    const int t  = threadIdx.x;
    const int n0 = blockIdx.y * 16;

    // Stage W slab (256K x 16N) into LDS as bf16, [n][k] layout.
    for (int i = t; i < 16 * DD; i += 256) {
        int n = i >> 8;          // 0..15
        int k = i & 255;         // 0..255
        float w = transposeW ? W[(long)(n0 + n) * DD + k]
                             : W[(long)k * DD + (n0 + n)];
        lw[n * LDS_STRIDE + k] = (__bf16)w;
    }
    __syncthreads();

    const int wave = t >> 5;
    const int lane = t & 31;
    const int half = lane >> 4;      // 0 or 1
    const int mloc = lane & 15;

    const int mtile = blockIdx.x * 8 + wave;
    const int mg    = mtile * 16 + mloc;
    const int mcl   = (mg < M) ? mg : 0;          // clamp: keep EXEC uniform for WMMA

    long rsrc = mcl;
    if (row_idx) {
        int r = row_idx[mcl];
        if (r < 0) r = 0;            // unset row: computed garbage fixed up afterwards
        rsrc = r;
    }
    const float* __restrict__ arow = A + rsrc * (long)DD;

    v8f acc = {};
    #pragma unroll
    for (int kb = 0; kb < DD; kb += 32) {
        // A fragment: lane (half,mloc) holds row M=mloc, K = kb + half*16 + {0..15}
        const float4* ap4 = (const float4*)(arow + kb + half * 16);
        float4 a0 = ap4[0], a1 = ap4[1], a2 = ap4[2], a3 = ap4[3];
        v16bf af;
        af[0]=(__bf16)a0.x;  af[1]=(__bf16)a0.y;  af[2]=(__bf16)a0.z;  af[3]=(__bf16)a0.w;
        af[4]=(__bf16)a1.x;  af[5]=(__bf16)a1.y;  af[6]=(__bf16)a1.z;  af[7]=(__bf16)a1.w;
        af[8]=(__bf16)a2.x;  af[9]=(__bf16)a2.y;  af[10]=(__bf16)a2.z; af[11]=(__bf16)a2.w;
        af[12]=(__bf16)a3.x; af[13]=(__bf16)a3.y; af[14]=(__bf16)a3.z; af[15]=(__bf16)a3.w;

        // B fragment: lane holds N=mloc, K = kb + half*16 + j (16 contiguous bf16 in LDS)
        v16bf bfr = *(const v16bf*)(&lw[mloc * LDS_STRIDE + kb + half * 16]);

        acc = __builtin_amdgcn_wmma_f32_16x16x32_bf16(
            /*neg_a=*/false, af, /*neg_b=*/false, bfr,
            /*c_mod=*/(short)0, acc, /*reuse_a=*/false, /*reuse_b=*/false);
    }

    // D layout: element r -> row (mtile*16 + r + 8*half), col n0 + mloc
    #pragma unroll
    for (int r = 0; r < 8; ++r) {
        int mrow = mtile * 16 + r + 8 * half;
        if (mrow < M) C[(long)mrow * DD + n0 + mloc] = acc[r];
    }
}

// ---------------- pass A: per-edge score + segment max (wave per edge) ----------------
__global__ void score_max_kernel(const float* __restrict__ qk, const float* __restrict__ enc,
                                 const int* __restrict__ ekey, const int* __restrict__ eidx,
                                 float* __restrict__ score, int* __restrict__ segmax, int E) {
    int g = blockIdx.x * blockDim.x + threadIdx.x;
    int e = g >> 5, lane = g & 31;
    if (e >= E) return;
    int seg = eidx[e];
    const float* __restrict__ q = qk  + (long)seg * DD;
    const float* __restrict__ x = enc + (long)ekey[e] * DD;
    float s = 0.0f;
    #pragma unroll
    for (int i = 0; i < 8; ++i) { int c = lane + 32 * i; s += q[c] * x[c]; }
    #pragma unroll
    for (int off = 16; off > 0; off >>= 1) s += __shfl_xor(s, off, 32);
    if (lane == 0) {
        s *= 0.0625f;   // 1/sqrt(256)
        score[e] = s;
        // sign-split integer trick == float atomic max (segmax inited to -inf bits)
        if (s >= 0.0f) atomicMax(segmax + seg, __float_as_int(s));
        else           atomicMin((unsigned int*)(segmax + seg), (unsigned int)__float_as_int(s));
    }
}

// ---------------- pass B: e = exp(s - max), segment denom ----------------
__global__ void exp_denom_kernel(float* __restrict__ score, const int* __restrict__ eidx,
                                 const int* __restrict__ segmax, float* __restrict__ denom, int E) {
    int e = blockIdx.x * blockDim.x + threadIdx.x;
    if (e >= E) return;
    int seg = eidx[e];
    float m  = __int_as_float(segmax[seg]);
    float ev = __expf(score[e] - m);
    score[e] = ev;
    atomicAdd(denom + seg, ev);
}

// ---------------- pass C: acc[seg] += prob * enc[key]  (wave per edge) ----------------
__global__ void scatter_acc_kernel(const float* __restrict__ score, const float* __restrict__ denom,
                                   const float* __restrict__ enc, const int* __restrict__ ekey,
                                   const int* __restrict__ eidx, float* __restrict__ acc, int E) {
    int g = blockIdx.x * blockDim.x + threadIdx.x;
    int e = g >> 5, lane = g & 31;
    if (e >= E) return;
    int seg = eidx[e];
    float prob = score[e] / denom[seg];
    const float* __restrict__ x = enc + (long)ekey[e] * DD;
    float* __restrict__ a = acc + (long)seg * DD;
    #pragma unroll
    for (int i = 0; i < 8; ++i) { int c = lane + 32 * i; atomicAdd(a + c, prob * x[c]); }
}

// ---------------- launcher ----------------
extern "C" void kernel_launch(void* const* d_in, const int* in_sizes, int n_in,
                              void* d_out, int out_size, void* d_ws, size_t ws_size,
                              hipStream_t stream) {
    const float* enc     = (const float*)d_in[0];
    const float* Wk      = (const float*)d_in[1];
    const float* Wv      = (const float*)d_in[2];
    const int*   a2p_key = (const int*)d_in[3];
    const int*   a2p_val = (const int*)d_in[4];
    const int*   p2r_key = (const int*)d_in[5];
    const int*   p2r_val = (const int*)d_in[6];

    const int E = in_sizes[4];   // edges (= N_AST)
    const int P = in_sizes[5];   // PDG nodes (= nr_cfg_nodes)

    // workspace layout
    char*  w      = (char*)d_ws;
    float* qk     = (float*)w;  w += (size_t)P * DD * sizeof(float);
    float* acc    = (float*)w;  w += (size_t)P * DD * sizeof(float);
    float* score  = (float*)w;  w += (size_t)E * sizeof(float);
    float* denom  = (float*)w;  w += (size_t)P * sizeof(float);
    int*   segmax = (int*)w;    w += (size_t)P * sizeof(int);
    int*   qidx   = (int*)w;    w += (size_t)P * sizeof(int);

    const long accElems   = (long)P * DD;
    const int  initBlocks = (int)((accElems + 255) / 256);
    init_ws_kernel<<<initBlocks, 256, 0, stream>>>(acc, denom, (unsigned*)segmax, qidx, P);

    scatter_qidx_kernel<<<(P + 255) / 256, 256, 0, stream>>>(p2r_key, p2r_val, qidx, P);

    // qk = enc[qidx] @ Wk^T
    dim3 ggrid((P + 127) / 128, 16);
    gemm_bf16_wmma_kernel<<<ggrid, 256, 0, stream>>>(enc, qidx, Wk, qk, P, /*transposeW=*/1);

    // zero any rows whose query was never scattered (generality; no-op for arange keys)
    fix_unset_rows_kernel<<<initBlocks, 256, 0, stream>>>(qk, qidx, P);

    // per-edge score + segment max   (wave per edge)
    long sThreads = (long)E * 32;
    score_max_kernel<<<(int)((sThreads + 255) / 256), 256, 0, stream>>>(
        qk, enc, a2p_key, a2p_val, score, segmax, E);

    exp_denom_kernel<<<(E + 255) / 256, 256, 0, stream>>>(score, a2p_val, segmax, denom, E);

    scatter_acc_kernel<<<(int)((sThreads + 255) / 256), 256, 0, stream>>>(
        score, denom, enc, a2p_key, a2p_val, acc, E);

    // out = acc @ Wv
    gemm_bf16_wmma_kernel<<<ggrid, 256, 0, stream>>>(acc, nullptr, Wv, (float*)d_out, P, /*transposeW=*/0);
}
